// ASAP_58033598104039
// MI455X (gfx1250) — matrix operand order repaired and checked by reading, not compiled, not run on hardware
//
#include <hip/hip_runtime.h>
#include <math.h>

#define B_   8
#define NPG  384
#define DEG  16
#define CIN  32
#define HID  64
#define NEF  8
#define NCLS 10
#define NTOT (B_*NPG)     /* 3072  */
#define ETOT (NTOT*DEG)   /* 49152 */

typedef __attribute__((ext_vector_type(16))) _Float16 v16h;
typedef __attribute__((ext_vector_type(8)))  float    v8f;

static inline int cdiv(int a, int b) { return (a + b - 1) / b; }

__device__ inline float eluf(float x) { return x > 0.f ? x : (expf(x) - 1.f); }

// ---------------------------------------------------------------------------
// Fragment loaders (A 16x32 f16 / B 32x16 f16, contiguous k = 16*half + s in both).
// Base pointers are pre-offset by the half-wave K sub-block (akb/bkb), so the
// remaining per-element offsets are the constants {0,2,4,6,8,10,12,14} (A)
// and {0..15} (B) -> loop-invariant scalar address math, no per-element mul.
// MASK versions are used only in the K%32 tail (rem already akb/bkb-adjusted).
// ---------------------------------------------------------------------------
template<bool CONTIG, bool MASK>
__device__ inline v16h load_afrag(const float* __restrict__ ApB, long csA, int remA)
{
    const int offs[8] = {0, 2, 4, 6, 8, 10, 12, 14};
    v16h af;
#pragma unroll
    for (int r = 0; r < 8; ++r) {
        int o = offs[r];
        float a0, a1;
        if (CONTIG) {
            float2 p = *(const float2*)(ApB + o);
            a0 = p.x; a1 = p.y;
        } else {
            a0 = ApB[(long)o * csA];
            a1 = ApB[(long)(o + 1) * csA];
        }
        if (MASK) {
            a0 = (o     < remA) ? a0 : 0.f;
            a1 = (o + 1 < remA) ? a1 : 0.f;
        }
        af[2 * r]     = (_Float16)a0;
        af[2 * r + 1] = (_Float16)a1;
    }
    return af;
}

template<bool MASK>
__device__ inline v16h load_bfrag(const float* __restrict__ BpB, long rsB, int remB)
{
    v16h bf;
#pragma unroll
    for (int r = 0; r < 8; ++r) {
        int o = 2 * r;
        float b0 = BpB[(long)o * rsB];
        float b1 = BpB[(long)(o + 1) * rsB];
        if (MASK) {
            b0 = (o     < remB) ? b0 : 0.f;
            b1 = (o + 1 < remB) ? b1 : 0.f;
        }
        bf[2 * r]     = (_Float16)b0;
        bf[2 * r + 1] = (_Float16)b1;
    }
    return bf;
}

// ---------------------------------------------------------------------------
// Generic batched GEMM on WMMA f16 (fp32 accumulate).
//   C[b] = act( rowscale[b] * (A[b] @ Bm[b]) + add[b] + bias )
// A accessed as A[m*rsA + k*csA]; CONTIG template = (csA==1) specialization.
// One wave per 16x64 output strip: A fragment reused across 4 WMMAs.
//
// Invariants exploited (all operands NPG- or exact-padded):
//  * every tile address (rows < ceil(M/16)*16, k < ceil(K/32)*32,
//    cols < ceil(N/64)*64) is in-bounds memory -> UNCONDITIONAL loads;
//  * A-row m only feeds D-row m, B-col n only D-col n; out-of-range rows/cols
//    are never stored, so junk there is self-contained;
//  * only the K tail needs value masking -> separate masked epilogue block.
// ---------------------------------------------------------------------------
template<bool CONTIG>
__global__ __launch_bounds__(32) void k_gemm_wmma(
    const float* __restrict__ A, long rsA, long csA, long bsA,
    const float* __restrict__ Bm, long rsB, long bsB,
    const float* __restrict__ bias,
    const float* __restrict__ add, long rsAdd, long bsAdd,
    const float* __restrict__ rowscale, long bsRS,
    float* __restrict__ C, long rsC, long bsC,
    int M, int N, int K, int act)
{
    int lane = threadIdx.x;
    int bm  = blockIdx.x * 16;
    int bn0 = blockIdx.y * 64;
    long bz = blockIdx.z;
    A  += bz * bsA;
    Bm += bz * bsB;
    C  += bz * bsC;
    if (add)      add      += bz * bsAdd;
    if (rowscale) rowscale += bz * bsRS;

    int half = lane >> 4;      // 0 | 1
    int l16  = lane & 15;
    int arow = bm + l16;       // A fragment row (padded-safe)
    int akb  = half * 16;      // K sub-block base for A
    int bkb  = half * 16;      // K sub-block base for B

    const float* ApB = A + (long)arow * rsA + (CONTIG ? (long)akb : (long)akb * csA);
    const float* BpB = Bm + bn0 + l16 + (long)bkb * rsB;

    v8f acc[4];
#pragma unroll
    for (int t = 0; t < 4; ++t) acc[t] = (v8f){0.f,0.f,0.f,0.f,0.f,0.f,0.f,0.f};

    const int kmain = K & ~31;
    int k0 = 0;
    for (; k0 < kmain; k0 += 32) {
        v16h af = load_afrag<CONTIG, false>(ApB, csA, 0);
#pragma unroll
        for (int t = 0; t < 4; ++t) {
            v16h bf = load_bfrag<false>(BpB + 16 * t, rsB, 0);
            acc[t] = __builtin_amdgcn_wmma_f32_16x16x32_f16(
                false, af, false, bf, (short)0, acc[t], false, false);
        }
        ApB += CONTIG ? 32 : 32 * csA;
        BpB += 32 * rsB;
    }
    if (k0 < K) {   // masked tail, runs at most once (K % 32 != 0)
        int remA = K - k0 - akb;
        int remB = K - k0 - bkb;
        v16h af = load_afrag<CONTIG, true>(ApB, csA, remA);
#pragma unroll
        for (int t = 0; t < 4; ++t) {
            v16h bf = load_bfrag<true>(BpB + 16 * t, rsB, remB);
            acc[t] = __builtin_amdgcn_wmma_f32_16x16x32_f16(
                false, af, false, bf, (short)0, acc[t], false, false);
        }
    }

#pragma unroll
    for (int t = 0; t < 4; ++t) {
#pragma unroll
        for (int r = 0; r < 8; ++r) {
            int m = bm + r + half * 8;   // C/D layout: VGPR r -> row r / r+8
            int n = bn0 + 16 * t + l16;
            if (m < M && n < N) {
                float v = acc[t][r];
                if (rowscale) v *= rowscale[m];
                if (add)      v += add[(long)m * rsAdd + n];
                if (bias)     v += bias[n];
                if (act == 1) v = eluf(v);
                C[(long)m * rsC + n] = v;
            }
        }
    }
}

// ---------------------------------------------------------------------------
// Elementwise / scatter / reduction helpers
// ---------------------------------------------------------------------------
__global__ void k_zero(float* p, long n) {
    long i = blockIdx.x * (long)blockDim.x + threadIdx.x;
    long st = (long)gridDim.x * blockDim.x;
    for (; i < n; i += st) p[i] = 0.f;
}

__global__ void k_edge_mlp(const float* __restrict__ ea,
                           const float* __restrict__ w1, const float* __restrict__ b1,
                           const float* __restrict__ w2, const float* __restrict__ b2,
                           const int* __restrict__ dst, float* __restrict__ h0)
{
    int e = blockIdx.x * blockDim.x + threadIdx.x;
    if (e >= ETOT) return;
    const float* a = ea + (long)e * NEF;
    float hid[20];
    for (int j = 0; j < 20; ++j) {
        float s = b1[j];
        for (int i = 0; i < NEF; ++i) s += a[i] * w1[i * 20 + j];
        hid[j] = eluf(s);
    }
    int d = dst[e];
    for (int c = 0; c < CIN; ++c) {
        float s = b2[c];
        for (int j = 0; j < 20; ++j) s += hid[j] * w2[j * CIN + c];
        atomicAdd(&h0[(long)d * CIN + c], s);
    }
}

__global__ void k_build_m0(const float* __restrict__ h0, const int* __restrict__ src,
                           const int* __restrict__ dst, float* __restrict__ m0)
{
    long t = blockIdx.x * (long)blockDim.x + threadIdx.x;
    if (t >= (long)ETOT * 64) return;
    int e = (int)(t >> 6), c = (int)(t & 63);
    int s = src[e], d = dst[e];
    float v = (c < CIN) ? h0[(long)d * CIN + c]
                        : (h0[(long)s * CIN + (c - CIN)] - h0[(long)d * CIN + (c - CIN)]);
    m0[t] = v;
}

__global__ void k_reduce2(const float* __restrict__ x, long n, float* red) {
    __shared__ float s0[256], s1[256];
    long i = blockIdx.x * (long)blockDim.x + threadIdx.x;
    long st = (long)gridDim.x * blockDim.x;
    float a = 0.f, b = 0.f;
    for (; i < n; i += st) { float v = x[i]; a += v; b += v * v; }
    s0[threadIdx.x] = a; s1[threadIdx.x] = b;
    __syncthreads();
    for (int o = 128; o > 0; o >>= 1) {
        if ((int)threadIdx.x < o) { s0[threadIdx.x] += s0[threadIdx.x + o]; s1[threadIdx.x] += s1[threadIdx.x + o]; }
        __syncthreads();
    }
    if (threadIdx.x == 0) { atomicAdd(&red[0], s0[0]); atomicAdd(&red[1], s1[0]); }
}

__global__ void k_apply_ln(float* x, long n, const float* __restrict__ red, float invn) {
    float m = red[0] * invn;
    float v = red[1] * invn - m * m;
    float s = rsqrtf(v);                       // reference has no eps
    long i = blockIdx.x * (long)blockDim.x + threadIdx.x;
    long st = (long)gridDim.x * blockDim.x;
    for (; i < n; i += st) x[i] = (x[i] - m) * s;
}

// out[dst[e]] += rows[gidx ? gidx[e] : e]   (nc columns)
__global__ void k_scatter_add(const float* __restrict__ rows, const int* __restrict__ gidx,
                              const int* __restrict__ dst, float* __restrict__ out, int nc)
{
    long t = blockIdx.x * (long)blockDim.x + threadIdx.x;
    if (t >= (long)ETOT * nc) return;
    int e = (int)(t / nc), c = (int)(t - (long)e * nc);
    int r = gidx ? gidx[e] : e;
    atomicAdd(&out[(long)dst[e] * nc + c], rows[(long)r * nc + c]);
}

__global__ void k_elu_ip(float* x, long n) {
    long i = blockIdx.x * (long)blockDim.x + threadIdx.x;
    long st = (long)gridDim.x * blockDim.x;
    for (; i < n; i += st) x[i] = eluf(x[i]);
}

__global__ void k_feast_fin(const float* __restrict__ agg, const float* __restrict__ lft,
                            const float* __restrict__ bias, float* __restrict__ X)
{
    long t = blockIdx.x * (long)blockDim.x + threadIdx.x;
    if (t >= (long)NTOT * HID) return;
    int c = (int)(t & 63);
    X[t] = eluf((agg[t] + lft[t]) * (1.f / 17.f) + bias[c]);  // in-degree 16 + self loop
}

__global__ void k_build_A(const int* __restrict__ src, const int* __restrict__ dst,
                          float* __restrict__ A)
{
    int e = blockIdx.x * blockDim.x + threadIdx.x;
    if (e >= ETOT) return;
    int s = src[e], d = dst[e];
    int g = s / NPG;
    A[(long)g * NPG * NPG + (long)(s % NPG) * NPG + (d % NPG)] = 1.f;
}

__global__ void k_colmean(const float* __restrict__ X, int k, float* __restrict__ out, int ostride) {
    int t = blockIdx.x * blockDim.x + threadIdx.x;
    if (t >= B_ * HID) return;
    int b = t / HID, c = t % HID;
    const float* Xb = X + (long)b * NPG * HID;
    float s = 0.f;
    for (int j = 0; j < k; ++j) s += Xb[(long)j * HID + c];
    out[(long)b * ostride + c] = s / (float)k;
}

__global__ void k_invcnt(const float* __restrict__ A, int k, float* __restrict__ out) {
    int t = blockIdx.x * blockDim.x + threadIdx.x;
    if (t >= B_ * k) return;
    int b = t / k, d = t % k;
    const float* Ab = A + (long)b * NPG * NPG;
    int c = 0;
    for (int s = 0; s < k; ++s) c += (Ab[(long)s * NPG + d] != 0.f) ? 1 : 0;
    out[(long)b * NPG + d] = 1.f / (float)(c > 0 ? c : 1);
}

__global__ void k_add_diag(float* A, int k) {
    int t = blockIdx.x * blockDim.x + threadIdx.x;
    if (t >= B_ * k) return;
    int b = t / k, d = t % k;
    A[(long)b * NPG * NPG + (long)d * NPG + d] += 1.f;
}

__global__ void k_xq(const float* __restrict__ A, const float* __restrict__ X, int k,
                     float* __restrict__ xq)
{
    long t = blockIdx.x * (long)blockDim.x + threadIdx.x;
    if (t >= (long)B_ * k * HID) return;
    int b = (int)(t / ((long)k * HID));
    long r = t - (long)b * k * HID;
    int d = (int)(r / HID), c = (int)(r % HID);
    const float* Ab = A + (long)b * NPG * NPG;
    const float* Xb = X + (long)b * NPG * HID;
    float m = -3.402823466e38f;
    for (int s = 0; s < k; ++s)
        if (Ab[(long)s * NPG + d] != 0.f) {
            float v = Xb[(long)s * HID + c];
            m = v > m ? v : m;
        }
    xq[(long)b * NPG * HID + (long)d * HID + c] = m;
}

__global__ void k_uq(const float* __restrict__ xq, const float* __restrict__ linw,
                     const float* __restrict__ linb, const float* __restrict__ wq,
                     float* __restrict__ uq, int k)
{
    int t = blockIdx.x * blockDim.x + threadIdx.x;
    if (t >= B_ * k) return;
    int b = t / k, d = t % k;
    const float* xr = xq + (long)b * NPG * HID + (long)d * HID;
    float acc = 0.f;
    for (int h = 0; h < HID; ++h) {
        float s = linb[h];
        for (int c = 0; c < HID; ++c) s += xr[c] * linw[c * HID + h];
        acc += s * wq[h];
    }
    uq[(long)b * NPG + d] = acc;
}

__global__ void k_vj(const float* __restrict__ X, const float* __restrict__ wj,
                     float* __restrict__ vj, int k)
{
    int t = blockIdx.x * blockDim.x + threadIdx.x;
    if (t >= B_ * k) return;
    int b = t / k, s = t % k;
    const float* xr = X + (long)b * NPG * HID + (long)s * HID;
    float acc = 0.f;
    for (int c = 0; c < HID; ++c) acc += xr[c] * wj[c];
    vj[(long)b * NPG + s] = acc;
}

__global__ void k_softmax_S(const float* __restrict__ A, const float* __restrict__ vj,
                            const float* __restrict__ uq, const float* __restrict__ attb,
                            float* __restrict__ S, int k)
{
    int t = blockIdx.x * blockDim.x + threadIdx.x;
    if (t >= B_ * k) return;
    int b = t / k, d = t % k;
    const float* Ab = A + (long)b * NPG * NPG;
    float uqd = uq[(long)b * NPG + d] + attb[0];
    float mx = -3.402823466e38f;
    for (int s = 0; s < k; ++s)
        if (Ab[(long)s * NPG + d] != 0.f) {
            float x = vj[(long)b * NPG + s] + uqd;
            float sc = x > 0.f ? x : 0.2f * x;       // leaky_relu(.,0.2)
            mx = sc > mx ? sc : mx;
        }
    float sum = 0.f;
    for (int s = 0; s < k; ++s)
        if (Ab[(long)s * NPG + d] != 0.f) {
            float x = vj[(long)b * NPG + s] + uqd;
            float sc = x > 0.f ? x : 0.2f * x;
            sum += expf(sc - mx);
        }
    float inv = 1.f / sum;
    float* Sb = S + (long)b * NPG * NPG;
    for (int s = 0; s < k; ++s) {
        float v = 0.f;
        if (Ab[(long)s * NPG + d] != 0.f) {
            float x = vj[(long)b * NPG + s] + uqd;
            float sc = x > 0.f ? x : 0.2f * x;
            v = expf(sc - mx) * inv;
        }
        Sb[(long)s * NPG + d] = v;
    }
}

__global__ void k_bb(const float* __restrict__ Xc, const float* __restrict__ le2w,
                     float* __restrict__ bb, int k)
{
    int t = blockIdx.x * blockDim.x + threadIdx.x;
    if (t >= B_ * k) return;
    int b = t / k, s = t % k;
    const float* xr = Xc + (long)b * NPG * HID + (long)s * HID;
    float acc = 0.f;
    for (int c = 0; c < HID; ++c) acc += xr[c] * le2w[c];
    bb[(long)b * NPG + s] = acc;
}

__global__ void k_fit(const float* __restrict__ Xc,
                      const float* __restrict__ le1w, const float* __restrict__ le1b,
                      const float* __restrict__ le3w, const float* __restrict__ le3b,
                      const float* __restrict__ A, const float* __restrict__ bb,
                      float* __restrict__ fit, int k)
{
    int t = blockIdx.x * blockDim.x + threadIdx.x;
    if (t >= B_ * k) return;
    int b = t / k, d = t % k;
    const float* xr = Xc + (long)b * NPG * HID + (long)d * HID;
    float a = le1b[0], l3 = le3b[0];
    for (int c = 0; c < HID; ++c) { a += xr[c] * le1w[c]; l3 += xr[c] * le3w[c]; }
    const float* Ab = A + (long)b * NPG * NPG;
    float deg = 0.f, md = 0.f;
    for (int s = 0; s < k; ++s)
        if (Ab[(long)s * NPG + d] != 0.f) { deg += 1.f; md += bb[(long)b * NPG + s]; }
    float z = deg * a - md + l3;
    fit[(long)b * NPG + d] = 1.f / (1.f + expf(-z));
}

__global__ void k_topk(const float* __restrict__ fit, int k, int kn,
                       int* __restrict__ idx, float* __restrict__ topv)
{
    __shared__ float sf[NPG];
    __shared__ int   used[NPG];
    int b = blockIdx.x;
    if (threadIdx.x != 0) return;
    for (int j = 0; j < k; ++j) { sf[j] = fit[(long)b * NPG + j]; used[j] = 0; }
    for (int t = 0; t < kn; ++t) {
        int bi = 0; float bv = -3.402823466e38f;
        for (int j = 0; j < k; ++j)
            if (!used[j] && sf[j] > bv) { bv = sf[j]; bi = j; }
        used[bi] = 1;
        idx[(long)b * NPG + t]  = bi;
        topv[(long)b * NPG + t] = bv;
    }
}

__global__ void k_gather_X(const float* __restrict__ Xc, const int* __restrict__ idx,
                           const float* __restrict__ topv, float* __restrict__ Xn, int kn)
{
    long t = blockIdx.x * (long)blockDim.x + threadIdx.x;
    if (t >= (long)B_ * kn * HID) return;
    int b = (int)(t / ((long)kn * HID));
    long r = t - (long)b * kn * HID;
    int p = (int)(r / HID), c = (int)(r % HID);
    int j = idx[(long)b * NPG + p];
    Xn[(long)b * NPG * HID + (long)p * HID + c] =
        Xc[(long)b * NPG * HID + (long)j * HID + c] * topv[(long)b * NPG + p];
}

__global__ void k_gather_S(const float* __restrict__ S, const int* __restrict__ idx,
                           float* __restrict__ Ssel, int k, int kn)
{
    long t = blockIdx.x * (long)blockDim.x + threadIdx.x;
    if (t >= (long)B_ * k * kn) return;
    int b = (int)(t / ((long)k * kn));
    long r = t - (long)b * k * kn;
    int s = (int)(r / kn), p = (int)(r % kn);
    int j = idx[(long)b * NPG + p];
    Ssel[(long)b * NPG * NPG + (long)s * NPG + p] =
        S[(long)b * NPG * NPG + (long)s * NPG + j];
}

__global__ void k_zero_diag(float* A, int kn) {
    int t = blockIdx.x * blockDim.x + threadIdx.x;
    if (t >= B_ * kn) return;
    int b = t / kn, p = t % kn;
    A[(long)b * NPG * NPG + (long)p * NPG + p] = 0.f;
}

__global__ void k_lin_small(const float* __restrict__ in, const float* __restrict__ w,
                            const float* __restrict__ bias, float* __restrict__ out,
                            int M, int K, int N, int act)
{
    int t = blockIdx.x * blockDim.x + threadIdx.x;
    if (t >= M * N) return;
    int m = t / N, n = t % N;
    float s = bias ? bias[n] : 0.f;
    for (int k = 0; k < K; ++k) s += in[(long)m * K + k] * w[(long)k * N + n];
    if (act == 1) s = eluf(s);
    out[(long)m * N + n] = s;
}

__global__ void k_log_softmax(const float* __restrict__ logits, float* __restrict__ out) {
    int b = blockIdx.x * blockDim.x + threadIdx.x;
    if (b >= B_) return;
    const float* r = logits + (long)b * NCLS;
    float mx = r[0];
    for (int j = 1; j < NCLS; ++j) mx = r[j] > mx ? r[j] : mx;
    float s = 0.f;
    for (int j = 0; j < NCLS; ++j) s += expf(r[j] - mx);
    float lse = mx + logf(s);
    for (int j = 0; j < NCLS; ++j) out[(long)b * NCLS + j] = r[j] - lse;
}

// ---------------------------------------------------------------------------
// Host orchestration
// ---------------------------------------------------------------------------
static void gemm(hipStream_t st,
                 const float* A, long rsA, long csA, long bsA,
                 const float* Bm, long rsB, long bsB,
                 const float* bias,
                 const float* add, long rsAdd, long bsAdd,
                 const float* rsc, long bsRS,
                 float* C, long rsC, long bsC,
                 int M, int N, int K, int act, int batch)
{
    dim3 g(cdiv(M, 16), cdiv(N, 64), batch);
    if (csA == 1)
        k_gemm_wmma<true><<<g, dim3(32), 0, st>>>(A, rsA, csA, bsA, Bm, rsB, bsB, bias,
                                                  add, rsAdd, bsAdd, rsc, bsRS,
                                                  C, rsC, bsC, M, N, K, act);
    else
        k_gemm_wmma<false><<<g, dim3(32), 0, st>>>(A, rsA, csA, bsA, Bm, rsB, bsB, bias,
                                                   add, rsAdd, bsAdd, rsc, bsRS,
                                                   C, rsC, bsC, M, N, K, act);
}

extern "C" void kernel_launch(void* const* d_in, const int* in_sizes, int n_in,
                              void* d_out, int out_size, void* d_ws, size_t ws_size,
                              hipStream_t stream)
{
    (void)in_sizes; (void)n_in; (void)out_size; (void)ws_size;
    // --- input pytree (insertion-order flattening of setup_inputs) ---
    auto F = [&](int i) { return (const float*)d_in[i]; };
    const float* edge_attr = F(1);
    const float *em_w1 = F(2), *em_b1 = F(3), *em_w2 = F(4), *em_b2 = F(5);
    const float* c1w[3] = { F(6), F(8), F(10) };
    const float* c1b[3] = { F(7), F(9), F(11) };
    const float *c2_linw = F(12), *c2_bias = F(15);
    // gconv i: rel.w, rel.b, root.w at 16+3i..
    // pool j: lin.w, lin.b, att.w, att.b, le1.w, le1.b, le2.w, le3.w, le3.b at 43+9j..
    const float *lin1w = F(88), *lin1b = F(89), *lin2w = F(90), *lin2b = F(91);
    const int* esrc = (const int*)d_in[92];
    const int* edst = (const int*)d_in[93];

    // --- workspace carving (floats); ~35 MB total, fits in L2 ---
    float* ws = (float*)d_ws;
    long o = 0;
    float* h0   = ws + o; o += (long)NTOT * CIN;
    float* hN   = ws + o; o += (long)NTOT * HID;
    float* XA   = ws + o; o += (long)B_ * NPG * HID;
    float* XB   = ws + o; o += (long)B_ * NPG * HID;
    float* W1   = ws + o; o += (long)B_ * NPG * HID;   // lft / M1 / xq
    float* W2   = ws + o; o += (long)B_ * NPG * HID;   // agg / Xroot / Xc
    float* bigE0 = ws + o; o += (long)ETOT * HID;       // m ping; later AA/AB alias
    float* bigE1 = ws + o; o += (long)ETOT * HID;       // m pong; later S/T alias
    float* Ssel  = ws + o; o += (long)B_ * NPG * NPG;
    float* invc  = ws + o; o += (long)B_ * NPG;
    float* uq    = ws + o; o += (long)B_ * NPG;
    float* vj    = ws + o; o += (long)B_ * NPG;
    float* bb    = ws + o; o += (long)B_ * NPG;
    float* fit   = ws + o; o += (long)B_ * NPG;
    float* topv  = ws + o; o += (long)B_ * NPG;
    int*   idx   = (int*)(ws + o); o += (long)B_ * NPG;
    float* zbuf  = ws + o; o += (long)B_ * 640;
    float* z2    = ws + o; o += (long)B_ * HID;
    float* lgt   = ws + o; o += (long)B_ * NCLS;
    float* red   = ws + o; o += 4;
    // aliases valid after the edge stage finishes:
    float* AA = bigE0;
    float* AB = bigE0 + (long)B_ * NPG * NPG;
    float* Sb = bigE1;
    float* Tb = bigE1 + (long)B_ * NPG * NPG;

    const long wsAb = (long)NPG * NPG;   // per-graph matrix batch stride
    const long wsXb = (long)NPG * HID;   // per-graph feature batch stride

    // ===== Stage 1: edge MLP scattered into h0 [N,32] =====
    k_zero<<<1024, 256, 0, stream>>>(h0, (long)NTOT * CIN);
    k_edge_mlp<<<cdiv(ETOT, 64), 64, 0, stream>>>(edge_attr, em_w1, em_b1, em_w2, em_b2, edst, h0);

    // ===== Stage 2: EdgeConv chain (WMMA GEMMs over E rows) =====
    k_build_m0<<<cdiv(ETOT * HID, 256), 256, 0, stream>>>(h0, esrc, edst, bigE0);
    const float invEn = 1.f / (float)((long)ETOT * HID);
    for (int j = 0; j < 3; ++j) {
        float* in  = (j & 1) ? bigE1 : bigE0;
        float* out = (j & 1) ? bigE0 : bigE1;
        gemm(stream, in, HID, 1, 0, c1w[j], HID, 0, c1b[j],
             nullptr, 0, 0, nullptr, 0, out, HID, 0, ETOT, HID, HID, /*elu*/1, 1);
        k_zero<<<1, 32, 0, stream>>>(red, 2);
        k_reduce2<<<1024, 256, 0, stream>>>(out, (long)ETOT * HID, red);
        k_apply_ln<<<1024, 256, 0, stream>>>(out, (long)ETOT * HID, red, invEn);
    }
    float* mfin = bigE1;   // after 3 blocks
    k_zero<<<1024, 256, 0, stream>>>(hN, (long)NTOT * HID);
    k_scatter_add<<<cdiv(ETOT * HID, 256), 256, 0, stream>>>(mfin, nullptr, edst, hN, HID);
    k_elu_ip<<<1024, 256, 0, stream>>>(hN, (long)NTOT * HID);

    // ===== Stage 3: FeaStConv (heads=1 => softmax==1; deg==17 by construction) =====
    gemm(stream, hN, HID, 1, 0, c2_linw, HID, 0, nullptr,
         nullptr, 0, 0, nullptr, 0, W1, HID, 0, NTOT, HID, HID, 0, 1);   // lft
    k_zero<<<1024, 256, 0, stream>>>(W2, (long)NTOT * HID);
    k_scatter_add<<<cdiv(ETOT * HID, 256), 256, 0, stream>>>(W1, esrc, edst, W2, HID);
    k_feast_fin<<<cdiv(NTOT * HID, 256), 256, 0, stream>>>(W2, W1, c2_bias, XA);

    // ===== Stage 4: densify adjacency =====
    k_zero<<<2048, 256, 0, stream>>>(AA, (long)B_ * NPG * NPG);
    k_build_A<<<cdiv(ETOT, 256), 256, 0, stream>>>(esrc, edst, AA);

    k_colmean<<<cdiv(B_ * HID, 256), 256, 0, stream>>>(XA, NPG, zbuf + 0, 640);

    // ===== Stage 5: 9 GraphConv layers + 4 ASAP poolings =====
    float *Xc = XA, *Xn = XB, *Ac = AA, *An = AB;
    int k = NPG, pj = 0;
    for (int i = 0; i < 9; ++i) {
        const float* relw  = F(16 + 3 * i);
        const float* relb  = F(17 + 3 * i);
        const float* rootw = F(18 + 3 * i);
        // inv in-count per dst
        k_invcnt<<<cdiv(B_ * k, 256), 256, 0, stream>>>(Ac, k, invc);
        // M1 = diag(invc) * (Ac^T @ X)          [k,64] per graph
        gemm(stream, Ac, 1, NPG, wsAb, Xc, HID, wsXb, nullptr,
             nullptr, 0, 0, invc, NPG, W1, HID, wsXb, k, HID, k, 0, B_);
        // Xroot = X @ root.w
        gemm(stream, Xc, HID, 1, wsXb, rootw, HID, 0, nullptr,
             nullptr, 0, 0, nullptr, 0, W2, HID, wsXb, k, HID, HID, 0, B_);
        // Xn = elu(M1 @ rel.w + rel.b + Xroot)
        gemm(stream, W1, HID, 1, wsXb, relw, HID, 0, relb,
             W2, HID, wsXb, nullptr, 0, Xn, HID, wsXb, k, HID, HID, 1, B_);
        { float* t = Xc; Xc = Xn; Xn = t; }
        k_colmean<<<cdiv(B_ * HID, 256), 256, 0, stream>>>(Xc, k, zbuf + (i + 1) * 64, 640);

        if ((i % 2 == 0) && i < 8) {
            const int pb = 43 + 9 * pj;
            const float *plinw = F(pb), *plinb = F(pb + 1), *pattw = F(pb + 2), *pattb = F(pb + 3);
            const float *ple1w = F(pb + 4), *ple1b = F(pb + 5), *ple2w = F(pb + 6);
            const float *ple3w = F(pb + 7), *ple3b = F(pb + 8);
            int kn = (9 * k + 9) / 10;           // ceil(0.9*k): 346,312,281,253

            k_add_diag<<<cdiv(B_ * k, 256), 256, 0, stream>>>(Ac, k);
            k_xq<<<cdiv(B_ * k * HID, 256), 256, 0, stream>>>(Ac, Xc, k, W1);
            k_uq<<<cdiv(B_ * k, 128), 128, 0, stream>>>(W1, plinw, plinb, pattw, uq, k);
            k_vj<<<cdiv(B_ * k, 256), 256, 0, stream>>>(Xc, pattw + HID, vj, k);
            k_softmax_S<<<cdiv(B_ * k, 128), 128, 0, stream>>>(Ac, vj, uq, pattb, Sb, k);
            // Xc_clusters = S^T @ X  -> W2
            gemm(stream, Sb, 1, NPG, wsAb, Xc, HID, wsXb, nullptr,
                 nullptr, 0, 0, nullptr, 0, W2, HID, wsXb, k, HID, k, 0, B_);
            k_bb<<<cdiv(B_ * k, 256), 256, 0, stream>>>(W2, ple2w, bb, k);
            k_fit<<<cdiv(B_ * k, 128), 128, 0, stream>>>(W2, ple1w, ple1b, ple3w, ple3b,
                                                         Ac, bb, fit, k);
            k_topk<<<B_, 32, 0, stream>>>(fit, k, kn, idx, topv);
            k_gather_X<<<cdiv(B_ * kn * HID, 256), 256, 0, stream>>>(W2, idx, topv, Xn, kn);
            k_gather_S<<<cdiv(B_ * k * kn, 256), 256, 0, stream>>>(Sb, idx, Ssel, k, kn);
            // T = A @ Ssel   [k,kn];  An = Ssel^T @ T  [kn,kn]
            gemm(stream, Ac, NPG, 1, wsAb, Ssel, NPG, wsAb, nullptr,
                 nullptr, 0, 0, nullptr, 0, Tb, NPG, wsAb, k, kn, k, 0, B_);
            gemm(stream, Ssel, 1, NPG, wsAb, Tb, NPG, wsAb, nullptr,
                 nullptr, 0, 0, nullptr, 0, An, NPG, wsAb, kn, kn, k, 0, B_);
            k_zero_diag<<<cdiv(B_ * kn, 256), 256, 0, stream>>>(An, kn);
            { float* t = Ac; Ac = An; An = t; }
            { float* t = Xc; Xc = Xn; Xn = t; }
            k = kn;
            ++pj;
        }
    }

    // ===== Stage 6: head =====
    k_lin_small<<<cdiv(B_ * HID, 256), 256, 0, stream>>>(zbuf, lin1w, lin1b, z2, B_, 640, HID, 1);
    k_lin_small<<<1, 128, 0, stream>>>(z2, lin2w, lin2b, lgt, B_, HID, NCLS, 0);
    k_log_softmax<<<1, 32, 0, stream>>>(lgt, (float*)d_out);
}
